// LSTMDecoder_80341658239147
// MI455X (gfx1250) — compile-verified, matrix operand order checked by reading, not run on hardware
//
#include <hip/hip_runtime.h>
#include <hip/hip_bf16.h>
#include <math.h>

// Dims: N=batch, T=steps, D=embed, H=hidden, L=enc locations, C=enc dim, V=vocab
#define NN 16
#define TT 256
#define DD 512
#define HH 1024
#define LL 196
#define CC 512
#define VV 32000

typedef __attribute__((ext_vector_type(16))) __bf16 v16bf;
typedef __attribute__((ext_vector_type(8)))  float  v8f;

#define BM 64
#define BN 128
#define BK 64
#define PAD 4   // bf16 rows: 8-byte aligned, spreads banks

#if __has_builtin(__builtin_amdgcn_tensor_load_to_lds)
#define USE_TDM 1
typedef __attribute__((ext_vector_type(4))) unsigned u32x4;
typedef __attribute__((ext_vector_type(8))) int      i32x8;
typedef __attribute__((ext_vector_type(4))) int      i32x4;
#else
#define USE_TDM 0
#endif

__device__ __forceinline__ __bf16 f2bf(float f) { return (__bf16)f; }

// Generic batched GEMM: C = act(A(MxK) @ B(KxN) + bias + Cadd)
//   bT: B stored transposed ((N x K) row-major, element (k,n) at B[n*ldb+k])
//   act: 0 none, 1 tanh, 2 LSTM-gate (col<gate_split -> sigmoid else tanh)
// f32 in/out, bf16 WMMA compute with f32 accumulate.
// A tile staged via Tensor Data Mover (f32, TDM row padding); B tile staged
// via VALU convert+transpose into bf16 LDS.
__global__ __launch_bounds__(256) void gemm_bf16_wmma(
    const float* __restrict__ A, const float* __restrict__ B, float* __restrict__ Cp,
    const float* __restrict__ bias, const float* __restrict__ Cadd,
    int M, int Nd, int K,
    int lda, int ldb, int ldc, int ldcadd,
    long long strideA, long long strideB, long long strideC,
    int bT, int act, int gate_split)
{
#if USE_TDM
    __shared__ float  sAf[BM][BK + 2];     // f32 A tile; row stride 66 dwords (TDM pad)
#else
    __shared__ __bf16 sA [BM][BK + PAD];   // bf16 A tile
#endif
    __shared__ __bf16 sBt[BN][BK + PAD];   // B tile TRANSPOSED, [n][k]

    const int tid  = threadIdx.x;
    const int wave = tid >> 5;
    const int lane = tid & 31;
    const int half = lane >> 4;     // K-half select
    const int l16  = lane & 15;     // M (A frag) / N (B,C frags)
    const int wm   = wave >> 2;     // 0..1  (2x4 wave grid over 64x128 tile)
    const int wn   = wave & 3;      // 0..3

    const int blockN = blockIdx.x * BN;
    const int blockM = blockIdx.y * BM;

    const float* Ab = A  + (long long)blockIdx.z * strideA;
    const float* Bb = B  + (long long)blockIdx.z * strideB;
    float*       Cb = Cp + (long long)blockIdx.z * strideC;

    const bool fullMN = (blockM + BM <= M) && (blockN + BN <= Nd);

    union AccU { v8f v; float e[8]; };
    AccU acc[2][2];
#pragma unroll
    for (int mi = 0; mi < 2; ++mi)
#pragma unroll
      for (int ni = 0; ni < 2; ++ni)
#pragma unroll
        for (int r = 0; r < 8; ++r) acc[mi][ni].e[r] = 0.0f;

    for (int kk = 0; kk < K; kk += BK) {
        const bool fast = fullMN && (kk + BK <= K);
        if (fast) {
#if USE_TDM
            // ---- Tensor Data Mover: DMA f32 A tile (BM x BK) global -> LDS ----
            // Issued by wave 0 only (TDM ignores EXEC; one DMA per workgroup).
            // D# group1 pad config inserts 2 dwords after every 64-dword row:
            // LDS row stride = 66 dwords = 264B (8B aligned, bank-conflict free).
            if (wave == 0) {
                unsigned long long ga =
                    (unsigned long long)(size_t)(Ab + (long long)blockM * lda + kk);
                unsigned ldsa = (unsigned)(size_t)&sAf[0][0];
                unsigned td0 = (unsigned)lda, td1 = (unsigned)M;
                u32x4 g0;
                g0[0] = 1u;                                    // count=1, user mode
                g0[1] = ldsa;                                  // lds_addr
                g0[2] = (unsigned)(ga & 0xFFFFFFFFull);        // global_addr[31:0]
                g0[3] = (unsigned)((ga >> 32) & 0x1FFFFFFull)  // global_addr[56:32]
                        | (2u << 30);                          // type=2 ("image")
                i32x8 g1 = {0, 0, 0, 0, 0, 0, 0, 0};
                g1[0] = (int)((2u << 16)       // data_size = 4 bytes
                            | (1u << 20)       // pad_enable
                            | (5u << 22)       // pad_interval: every 64 dwords
                            | (1u << 25));     // pad_amount: 2 dwords
                g1[1] = (int)((td0 & 0xFFFFu) << 16);                 // tensor_dim0 lo
                g1[2] = (int)((td0 >> 16) | ((td1 & 0xFFFFu) << 16)); // dim0 hi | dim1 lo
                g1[3] = (int)((td1 >> 16) | ((unsigned)BK << 16));    // dim1 hi | tile_dim0
                g1[4] = (int)BM;                                       // tile_dim1
                g1[5] = (int)(unsigned)lda;                            // tensor_dim0_stride
                i32x4 gz = {0, 0, 0, 0};
#if defined(__clang_major__) && (__clang_major__ >= 23)
                i32x8 gz8 = {0, 0, 0, 0, 0, 0, 0, 0};
                __builtin_amdgcn_tensor_load_to_lds(g0, g1, gz, gz, gz8, 0);
#else
                __builtin_amdgcn_tensor_load_to_lds(g0, g1, gz, gz, 0);
#endif
            }
#else
            // A: BM*BK = 4096 elems -> 4 x float4 per thread, convert, b64 store
#pragma unroll
            for (int i = 0; i < (BM * BK) / (256 * 4); ++i) {
                int idx4 = tid + i * 256;
                int r = idx4 >> 4, kq = (idx4 & 15) << 2;
                const float4 f = *(const float4*)&Ab[(long long)(blockM + r) * lda + kk + kq];
                __bf16 p[4] = { f2bf(f.x), f2bf(f.y), f2bf(f.z), f2bf(f.w) };
                *(uint2*)&sA[r][kq] = *(const uint2*)p;
            }
#endif
            if (bT) {
                // B^T in memory: along-k contiguous -> float4 + packed b64 store
#pragma unroll
                for (int i = 0; i < (BN * BK) / (256 * 4); ++i) {
                    int idx4 = tid + i * 256;
                    int n = idx4 >> 4, kq = (idx4 & 15) << 2;
                    const float4 f = *(const float4*)&Bb[(long long)(blockN + n) * ldb + kk + kq];
                    __bf16 p[4] = { f2bf(f.x), f2bf(f.y), f2bf(f.z), f2bf(f.w) };
                    *(uint2*)&sBt[n][kq] = *(const uint2*)p;
                }
            } else {
                // row-major B: coalesced float4 along n, transposed b16 stores
#pragma unroll
                for (int i = 0; i < (BN * BK) / (256 * 4); ++i) {
                    int idx4 = tid + i * 256;
                    int kr = idx4 >> 5, nq = (idx4 & 31) << 2;
                    const float4 f = *(const float4*)&Bb[(long long)(kk + kr) * ldb + blockN + nq];
                    sBt[nq + 0][kr] = f2bf(f.x);
                    sBt[nq + 1][kr] = f2bf(f.y);
                    sBt[nq + 2][kr] = f2bf(f.z);
                    sBt[nq + 3][kr] = f2bf(f.w);
                }
            }
#if USE_TDM
            if (wave == 0) {
#if __has_builtin(__builtin_amdgcn_s_wait_tensorcnt)
                __builtin_amdgcn_s_wait_tensorcnt(0);
#else
                asm volatile("s_wait_tensorcnt 0x0" ::: "memory");
#endif
            }
#endif
        } else {
            // ---------- ragged path: branchless clamped loads ----------
#pragma unroll
            for (int i = 0; i < (BM * BK) / 256; ++i) {
                int idx = tid + i * 256;
                int r = idx >> 6, k = idx & 63;
                int gm = blockM + r, gk = kk + k;
                int gmc = gm < M ? gm : M - 1;
                int gkc = gk < K ? gk : K - 1;
                float v = Ab[(long long)gmc * lda + gkc];
                v = (gm < M && gk < K) ? v : 0.0f;
#if USE_TDM
                sAf[r][k] = v;
#else
                sA[r][k] = f2bf(v);
#endif
            }
#pragma unroll
            for (int i = 0; i < (BN * BK) / 256; ++i) {
                int idx = tid + i * 256;
                int k = idx >> 7, n = idx & 127;
                int gn = blockN + n, gk = kk + k;
                int gnc = gn < Nd ? gn : Nd - 1;
                int gkc = gk < K ? gk : K - 1;
                float v = bT ? Bb[(long long)gnc * ldb + gkc]
                             : Bb[(long long)gkc * ldb + gnc];
                sBt[n][k] = f2bf((gn < Nd && gk < K) ? v : 0.0f);
            }
        }
        if (kk + BK < K)
            __builtin_prefetch(&Ab[(long long)(blockM + (tid >> 2)) * lda + kk + BK], 0, 1);
        __syncthreads();

        // WMMA fragment gathers. kv = (v<4 ? 2v : 16+2(v-4)) + 8*half.
        // A: f32 pair (ds_load_b64) -> packed bf16 convert. B: bf16 pair as one dword.
        union FragU { v16bf v; unsigned u[8]; };
#pragma unroll
        for (int ks = 0; ks < BK; ks += 32) {
            FragU af[2], bfr[2];
#pragma unroll
            for (int mi = 0; mi < 2; ++mi) {
                int row = wm * 32 + mi * 16 + l16;
#pragma unroll
                for (int v = 0; v < 8; ++v) {
                    int kv = ((v < 4) ? 2 * v : 16 + 2 * (v - 4)) + 8 * half;
#if USE_TDM
                    const float2 fa = *(const float2*)&sAf[row][ks + kv];
                    __bf16 pa[2] = { f2bf(fa.x), f2bf(fa.y) };
                    af[mi].u[v] = *(const unsigned*)pa;
#else
                    af[mi].u[v] = *(const unsigned*)&sA[row][ks + kv];
#endif
                }
            }
#pragma unroll
            for (int ni = 0; ni < 2; ++ni) {
                int col = wn * 32 + ni * 16 + l16;
#pragma unroll
                for (int v = 0; v < 8; ++v) {
                    int kv = ((v < 4) ? 2 * v : 16 + 2 * (v - 4)) + 8 * half;
                    bfr[ni].u[v] = *(const unsigned*)&sBt[col][ks + kv];
                }
            }
#pragma unroll
            for (int mi = 0; mi < 2; ++mi)
#pragma unroll
              for (int ni = 0; ni < 2; ++ni)
                acc[mi][ni].v = __builtin_amdgcn_wmma_f32_16x16x32_bf16(
                    false, af[mi].v, false, bfr[ni].v, (short)0, acc[mi][ni].v, false, false);
        }
        __syncthreads();
    }

    // Epilogue: bias + residual-add + activation, guarded stores.
#pragma unroll
    for (int mi = 0; mi < 2; ++mi) {
#pragma unroll
        for (int ni = 0; ni < 2; ++ni) {
            int col = blockN + wn * 32 + ni * 16 + l16;
            if (col >= Nd) continue;
            float bv = bias ? bias[col] : 0.0f;
#pragma unroll
            for (int r = 0; r < 8; ++r) {
                int row = blockM + wm * 32 + mi * 16 + half * 8 + r;
                if (row >= M) continue;
                float v = acc[mi][ni].e[r] + bv;
                if (Cadd) v += Cadd[(long long)row * ldcadd + col];
                if (act == 1)      v = tanhf(v);
                else if (act == 2) v = (col < gate_split) ? (1.0f / (1.0f + __expf(-v)))
                                                          : tanhf(v);
                Cb[(long long)row * ldc + col] = v;
            }
        }
    }
}

// Per-step LSTM pointwise: c2 = f*c + i*g ; h2 = o*tanh(c2); record h2 in hs[:,t,:]
__global__ void lstm_pointwise(const float* __restrict__ gates,
                               float* __restrict__ cbuf, float* __restrict__ hbuf,
                               float* __restrict__ hs, int t)
{
    int idx = blockIdx.x * blockDim.x + threadIdx.x;
    if (idx >= NN * HH) return;
    int n = idx / HH, j = idx % HH;
    const float* g = gates + (long long)n * 4 * HH;
    float ig = g[j], fg = g[HH + j], og = g[2 * HH + j], gg = g[3 * HH + j];
    float c = fg * cbuf[idx] + ig * gg;
    float h = og * tanhf(c);
    cbuf[idx] = c;
    hbuf[idx] = h;
    hs[((long long)n * TT + t) * HH + j] = h;
}

// One wave32 per row softmax (rows = N*T, cols = L)
__global__ void softmax_rows(const float* __restrict__ in, float* __restrict__ out, int cols)
{
    int row  = blockIdx.x;
    int lane = threadIdx.x;
    const float* r = in + (long long)row * cols;
    float m = -1e30f;
    for (int i = lane; i < cols; i += 32) m = fmaxf(m, r[i]);
#pragma unroll
    for (int o = 16; o > 0; o >>= 1) m = fmaxf(m, __shfl_xor(m, o, 32));
    float s = 0.0f;
    for (int i = lane; i < cols; i += 32) s += __expf(r[i] - m);
#pragma unroll
    for (int o = 16; o > 0; o >>= 1) s += __shfl_xor(s, o, 32);
    float inv = 1.0f / s;
    float* w = out + (long long)row * cols;
    for (int i = lane; i < cols; i += 32) w[i] = __expf(r[i] - m) * inv;
}

// Copy src(rows x width) into dst columns [dstoff, dstoff+width) with leading dim dstld
__global__ void copy_into_cols(const float* __restrict__ src, float* __restrict__ dst,
                               long long total, int width, int dstld, int dstoff)
{
    long long idx = (long long)blockIdx.x * blockDim.x + threadIdx.x;
    if (idx >= total) return;
    long long r = idx / width;
    int c = (int)(idx % width);
    dst[r * dstld + dstoff + c] = src[idx];
}

extern "C" void kernel_launch(void* const* d_in, const int* in_sizes, int n_in,
                              void* d_out, int out_size, void* d_ws, size_t ws_size,
                              hipStream_t stream)
{
    const float* X   = (const float*)d_in[0];   // (N,T,D)
    const float* h0  = (const float*)d_in[1];   // (N,1,H)
    const float* c0  = (const float*)d_in[2];   // (N,1,H)
    const float* enc = (const float*)d_in[3];   // (N,L,C)
    const float* Wx  = (const float*)d_in[4];   // (D,4H)
    const float* Wh  = (const float*)d_in[5];   // (H,4H)
    const float* bb  = (const float*)d_in[6];   // (4H,)
    const float* Wo  = (const float*)d_in[7];   // (H,V)
    const float* bo  = (const float*)d_in[8];   // (V,)
    const float* Wa  = (const float*)d_in[9];   // (H,C)
    const float* Wc  = (const float*)d_in[10];  // (C+H,H)

    float* out    = (float*)d_out;
    float* logits = out;                                    // (N,T,V)
    float* hT     = out + (size_t)NN * TT * VV;             // (N,1,H)
    float* cT     = hT  + (size_t)NN * HH;                  // (N,1,H)
    float* attn   = cT  + (size_t)NN * HH;                  // (N,T,L)

    // Workspace layout (floats); ~138 MB total
    float* ws = (float*)d_ws;
    size_t o = 0;
    float* xwx    = ws + o; o += (size_t)NN * TT * 4 * HH;  // x@Wx + b
    float* hs     = ws + o; o += (size_t)NN * TT * HH;      // all h_t
    float* hbuf   = ws + o; o += (size_t)NN * HH;
    float* cbuf   = ws + o; o += (size_t)NN * HH;
    float* gates  = ws + o; o += (size_t)NN * 4 * HH;
    float* proj   = ws + o; o += (size_t)NN * TT * CC;      // hs @ Wa
    float* scores = ws + o; o += (size_t)NN * TT * LL;
    float* concat = ws + o; o += (size_t)NN * TT * (CC + HH);
    float* attno  = ws + o; o += (size_t)NN * TT * HH;      // tanh(concat@Wc)

    auto gemm = [&](const float* A, const float* B, float* C,
                    const float* bias, const float* Cadd,
                    int M, int Nd, int K, int lda, int ldb, int ldc, int ldcadd,
                    long long sA, long long sB, long long sC, int batch,
                    int bT, int act, int gs) {
        dim3 grid((Nd + BN - 1) / BN, (M + BM - 1) / BM, batch);
        gemm_bf16_wmma<<<grid, 256, 0, stream>>>(A, B, C, bias, Cadd,
            M, Nd, K, lda, ldb, ldc, ldcadd, sA, sB, sC, bT, act, gs);
    };

    // init recurrent state
    hipMemcpyAsync(hbuf, h0, (size_t)NN * HH * sizeof(float), hipMemcpyDeviceToDevice, stream);
    hipMemcpyAsync(cbuf, c0, (size_t)NN * HH * sizeof(float), hipMemcpyDeviceToDevice, stream);

    // 1) Precompute x@Wx + b for all timesteps:  (N*T, D) @ (D, 4H)
    gemm(X, Wx, xwx, bb, nullptr,
         NN * TT, 4 * HH, DD, DD, 4 * HH, 4 * HH, 0, 0, 0, 0, 1, 0, 0, 0);

    // 2) Sequential LSTM: per step gates = h@Wh + xwx[:,t,:]; fused gate act in epilogue
    for (int t = 0; t < TT; ++t) {
        gemm(hbuf, Wh, gates, nullptr, xwx + (size_t)t * 4 * HH,
             NN, 4 * HH, HH, HH, 4 * HH, 4 * HH, TT * 4 * HH,
             0, 0, 0, 1, 0, /*act=*/2, /*gate_split=*/3 * HH);
        lstm_pointwise<<<(NN * HH + 255) / 256, 256, 0, stream>>>(gates, cbuf, hbuf, hs, t);
    }

    // 3) proj = hs @ Wa : (N*T, H) @ (H, C)
    gemm(hs, Wa, proj, nullptr, nullptr,
         NN * TT, CC, HH, HH, CC, CC, 0, 0, 0, 0, 1, 0, 0, 0);

    // 4) scores[n] = proj[n] @ enc[n]^T : batched (T,C)@(C,L), B transposed
    gemm(proj, enc, scores, nullptr, nullptr,
         TT, LL, CC, CC, CC, LL, 0,
         (long long)TT * CC, (long long)LL * CC, (long long)TT * LL, NN,
         /*bT=*/1, 0, 0);

    // 5) softmax over L -> attn output region
    softmax_rows<<<NN * TT, 32, 0, stream>>>(scores, attn, LL);

    // 6) context[n] = attn[n] @ enc[n] : (T,L)@(L,C), stored into concat cols [0,C)
    gemm(attn, enc, concat, nullptr, nullptr,
         TT, CC, LL, LL, CC, CC + HH, 0,
         (long long)TT * LL, (long long)LL * CC, (long long)TT * (CC + HH), NN,
         0, 0, 0);

    // 7) concat cols [C, C+H) = hs
    {
        long long tot = (long long)NN * TT * HH;
        copy_into_cols<<<(unsigned)((tot + 255) / 256), 256, 0, stream>>>(
            hs, concat, tot, HH, CC + HH, CC);
    }

    // 8) attno = tanh(concat @ Wc) : (N*T, C+H) @ (C+H, H)
    gemm(concat, Wc, attno, nullptr, nullptr,
         NN * TT, HH, CC + HH, CC + HH, HH, HH, 0, 0, 0, 0, 1, 0, /*act=*/1, 0);

    // 9) logits = attno @ Wo + bo : (N*T, H) @ (H, V)
    gemm(attno, Wo, logits, bo, nullptr,
         NN * TT, VV, HH, HH, VV, VV, 0, 0, 0, 0, 1, 0, 0, 0);

    // 10) final hT, cT
    hipMemcpyAsync(hT, hbuf, (size_t)NN * HH * sizeof(float), hipMemcpyDeviceToDevice, stream);
    hipMemcpyAsync(cT, cbuf, (size_t)NN * HH * sizeof(float), hipMemcpyDeviceToDevice, stream);
}
